// MyModel_87522843558524
// MI455X (gfx1250) — matrix operand order repaired and checked by reading, not compiled, not run on hardware
//
#include <hip/hip_runtime.h>

typedef __attribute__((ext_vector_type(16))) _Float16 v16h;
typedef __attribute__((ext_vector_type(8)))  float    v8f;

#define SEQ   80
#define EMB   100
#define UNITS 64
#define KPAD  128   // EMB padded to 4 x K=32 WMMA steps
#define ROWS  16    // batch rows per workgroup

// Branch-free tanh: tanh(x) = 1 - 2/(exp(2x)+1), via v_exp_f32 / v_rcp_f32.
__device__ __forceinline__ float fast_tanh(float x) {
  float e = __builtin_amdgcn_exp2f(x * 2.885390082f);   // exp2(2x*log2(e))
  return 1.0f - 2.0f * __builtin_amdgcn_rcpf(e + 1.0f);
}

// Branch-free sigmoid: 1/(1+exp(-x))
__device__ __forceinline__ float fast_sigmoid(float x) {
  float e = __builtin_amdgcn_exp2f(-x * 1.442695041f);
  return __builtin_amdgcn_rcpf(1.0f + e);
}

// Loop-invariant B-matrix fragment (32x16 f16) from row-major f32 W[K][64].
// 16-bit B layout: half e, lane L -> N = L%16, K = kb + 8*(L>>4) + e (e<8), kb + 16 + 8*(L>>4) + (e-8) (e>=8).
__device__ __forceinline__ v16h load_bfrag(const float* __restrict__ W, int kb,
                                           int krows, int ncol, int lhi) {
  v16h b;
#pragma unroll
  for (int e = 0; e < 16; ++e) {
    int k = kb + 8 * lhi + (e < 8 ? e : e + 8);
    float v = (k < krows) ? W[k * UNITS + ncol] : 0.0f;
    b[e] = (_Float16)v;
  }
  return b;
}

// A-matrix fragment (16x32 f16) from row-major f16 LDS tile, row stride `stride`.
// 16-bit A layout: lane L -> M = L%16; halves 0..7 at K = kb + e + 8*hi,
// halves 8..15 at K = kb + 16 + (e-8) + 8*hi  -> two contiguous 16-byte runs.
__device__ __forceinline__ v16h load_afrag(const _Float16* base, int stride,
                                           int kb, int lrow, int lhi) {
  const _Float16* p = base + lrow * stride + kb + 8 * lhi;
  v16h a;
#pragma unroll
  for (int e = 0; e < 8; ++e) { a[e] = p[e]; a[e + 8] = p[e + 16]; }
  return a;
}

__global__ __launch_bounds__(128) void rnn2_wmma_kernel(
    const int*   __restrict__ tokens, const float* __restrict__ emb,
    const float* __restrict__ Wx0,    const float* __restrict__ Wh0,
    const float* __restrict__ b0,     const float* __restrict__ Wx1,
    const float* __restrict__ Wh1,    const float* __restrict__ b1,
    const float* __restrict__ Wout,   const float* __restrict__ bout,
    float* __restrict__ out)
{
  __shared__ _Float16 xs [ROWS * KPAD];        // x_t tile (cols 100..127 stay 0)
  __shared__ _Float16 h0s[2][ROWS * UNITS];    // ping-pong layer-1 state (f16)
  __shared__ _Float16 h1s[2][ROWS * UNITS];    // ping-pong layer-2 state (f16)
  __shared__ int      toks[ROWS * SEQ];        // this block's token tile

  const int tid   = threadIdx.x;
  const int lane  = tid & 31;
  const int wave  = tid >> 5;       // 4 waves: each owns 16 of the 64 units
  const int lrow  = lane & 15;      // M row (A) / N col (B,C,D)
  const int lhi   = lane >> 4;      // half-wave selector
  const int ncol  = wave * 16 + lrow;
  const int brow0 = blockIdx.x * ROWS;

  // ---- one-time: weights f32 -> f16 WMMA B fragments in registers ----
  v16h bx0[4], bh0[2], bx1[2], bh1[2];
#pragma unroll
  for (int i = 0; i < 4; ++i) bx0[i] = load_bfrag(Wx0, i * 32, EMB,   ncol, lhi);
#pragma unroll
  for (int i = 0; i < 2; ++i) bh0[i] = load_bfrag(Wh0, i * 32, UNITS, ncol, lhi);
#pragma unroll
  for (int i = 0; i < 2; ++i) bx1[i] = load_bfrag(Wx1, i * 32, UNITS, ncol, lhi);
#pragma unroll
  for (int i = 0; i < 2; ++i) bh1[i] = load_bfrag(Wh1, i * 32, UNITS, ncol, lhi);

  const float bias0 = b0[ncol];
  const float bias1 = b1[ncol];

  // ---- one-time LDS init: zero xs (incl. K-pad) and initial h state;
  //      coalesced preload of the 16x80 token tile ----
  for (int i = tid; i < ROWS * KPAD; i += 128) xs[i] = (_Float16)0.0f;
  for (int i = tid; i < ROWS * UNITS; i += 128) {
    h0s[0][i] = (_Float16)0.0f; h1s[0][i] = (_Float16)0.0f;
  }
  for (int i = tid; i < ROWS * SEQ; i += 128) toks[i] = tokens[brow0 * SEQ + i];
  __syncthreads();

  for (int t = 0; t < SEQ; ++t) {
    const int cur = t & 1, nxt = cur ^ 1;

    // ---- gather x_t: 400 aligned float4 chunks, f32 -> f16, no branches ----
    for (int i = tid; i < ROWS * (EMB / 4); i += 128) {   // EMB/4 = 25 chunks/row
      const int row = i / 25;
      const int q   = i - row * 25;
      const int tok = toks[row * SEQ + t];
      const float4 v = ((const float4*)(emb + (long)tok * EMB))[q];
      _Float16* dst = &xs[row * KPAD + q * 4];
      dst[0] = (_Float16)v.x; dst[1] = (_Float16)v.y;
      dst[2] = (_Float16)v.z; dst[3] = (_Float16)v.w;
    }
    __syncthreads();

    // ---- layer 1: h0[nxt] = tanh(x Wx0 + h0[cur] Wh0 + b0) ----
    v16h ax[4], ah[2];
#pragma unroll
    for (int i = 0; i < 4; ++i) ax[i] = load_afrag(xs,       KPAD,  i * 32, lrow, lhi);
#pragma unroll
    for (int i = 0; i < 2; ++i) ah[i] = load_afrag(h0s[cur], UNITS, i * 32, lrow, lhi);

    v8f acc;
#pragma unroll
    for (int r = 0; r < 8; ++r) acc[r] = bias0;
#pragma unroll
    for (int i = 0; i < 4; ++i)
      acc = __builtin_amdgcn_wmma_f32_16x16x32_f16(false, ax[i], false, bx0[i],
                                                   (short)0, acc, false, false);
#pragma unroll
    for (int i = 0; i < 2; ++i)
      acc = __builtin_amdgcn_wmma_f32_16x16x32_f16(false, ah[i], false, bh0[i],
                                                   (short)0, acc, false, false);

    // C/D layout: VGPR r, lane -> (M = r + 8*lhi, N = lane%16)
#pragma unroll
    for (int r = 0; r < 8; ++r)
      h0s[nxt][(r + 8 * lhi) * UNITS + ncol] = (_Float16)fast_tanh(acc[r]);
    __syncthreads();

    // ---- layer 2: h1[nxt] = tanh(h0[nxt] Wx1 + h1[cur] Wh1 + b1) ----
    v16h a0[2], a1[2];
#pragma unroll
    for (int i = 0; i < 2; ++i) a0[i] = load_afrag(h0s[nxt], UNITS, i * 32, lrow, lhi);
#pragma unroll
    for (int i = 0; i < 2; ++i) a1[i] = load_afrag(h1s[cur], UNITS, i * 32, lrow, lhi);

#pragma unroll
    for (int r = 0; r < 8; ++r) acc[r] = bias1;
#pragma unroll
    for (int i = 0; i < 2; ++i)
      acc = __builtin_amdgcn_wmma_f32_16x16x32_f16(false, a0[i], false, bx1[i],
                                                   (short)0, acc, false, false);
#pragma unroll
    for (int i = 0; i < 2; ++i)
      acc = __builtin_amdgcn_wmma_f32_16x16x32_f16(false, a1[i], false, bh1[i],
                                                   (short)0, acc, false, false);

#pragma unroll
    for (int r = 0; r < 8; ++r)
      h1s[nxt][(r + 8 * lhi) * UNITS + ncol] = (_Float16)fast_tanh(acc[r]);
    __syncthreads();
  }

  // ---- head: sigmoid(h1 . Wout + bout), one thread per batch row ----
  // After SEQ (even) steps the final h1 lives in buffer SEQ & 1.
  if (tid < ROWS) {
    const _Float16* h = &h1s[SEQ & 1][tid * UNITS];
    float s = bout[0];
#pragma unroll
    for (int k = 0; k < UNITS; ++k) s += (float)h[k] * Wout[k];
    out[brow0 + tid] = fast_sigmoid(s);
  }
}

extern "C" void kernel_launch(void* const* d_in, const int* in_sizes, int n_in,
                              void* d_out, int out_size, void* d_ws, size_t ws_size,
                              hipStream_t stream) {
  const int*   tokens = (const int*)  d_in[0];
  const float* emb    = (const float*)d_in[1];
  const float* Wx0    = (const float*)d_in[2];
  const float* Wh0    = (const float*)d_in[3];
  const float* b0     = (const float*)d_in[4];
  const float* Wx1    = (const float*)d_in[5];
  const float* Wh1    = (const float*)d_in[6];
  const float* b1     = (const float*)d_in[7];
  const float* Wout   = (const float*)d_in[8];
  const float* bout   = (const float*)d_in[9];
  float* out = (float*)d_out;

  const int B = in_sizes[0] / SEQ;        // 16384
  const int nblocks = B / ROWS;           // 1024 workgroups x 128 threads
  rnn2_wmma_kernel<<<nblocks, 128, 0, stream>>>(tokens, emb, Wx0, Wh0, b0,
                                                Wx1, Wh1, b1, Wout, bout, out);
}
